// GConv_51307679318311
// MI455X (gfx1250) — matrix operand order repaired and checked by reading, not compiled, not run on hardware
//
#include <hip/hip_runtime.h>

#define DIM 128

typedef __attribute__((ext_vector_type(16))) __bf16 v16bf;
typedef __attribute__((ext_vector_type(8)))  float  v8f;

// ---------------------------------------------------------------------------
// deg[i] = 1 (self-loop weight); then atomic-accumulate edge weights by col.
// ---------------------------------------------------------------------------
__global__ void init_deg_kernel(float* deg, int n) {
    int i = blockIdx.x * blockDim.x + threadIdx.x;
    if (i < n) deg[i] = 1.0f;
}

__global__ void degree_kernel(const long long* col, const float* ew,
                              float* deg, int nedges) {
    int e = blockIdx.x * blockDim.x + threadIdx.x;
    if (e < nedges) atomicAdd(&deg[(int)col[e]], ew[e]);
}

__global__ void dinv_kernel(float* deg, int n) {
    int i = blockIdx.x * blockDim.x + threadIdx.x;
    if (i < n) {
        float d = deg[i];
        deg[i] = (d > 0.0f) ? rsqrtf(d) : 0.0f;
    }
}

__global__ void zero_kernel(float* p, int n) {
    int i = blockIdx.x * blockDim.x + threadIdx.x;
    if (i < n) p[i] = 0.0f;
}

// ---------------------------------------------------------------------------
// OUT[m, :] = H[m, :] @ W   (nrows x 128) * (128 x 128), f32 in/out,
// bf16 WMMA with f32 accumulation. 8 waves/block, 16 rows/wave, nrows % 16 == 0.
// W staged in LDS as bf16 column-major: WT[n*128 + k].
// ---------------------------------------------------------------------------
__global__ __launch_bounds__(256) void gemm_bf16_kernel(const float* __restrict__ H,
                                                        const float* __restrict__ W,
                                                        float* __restrict__ OUT,
                                                        int nrows) {
    __shared__ __align__(64) __bf16 WT[DIM * DIM];
    for (int i = threadIdx.x; i < DIM * DIM; i += 256) {
        int k = i >> 7, n = i & 127;              // W is row-major W[k][n]
        WT[n * DIM + k] = (__bf16)W[i];
    }
    __syncthreads();

    const int wave = threadIdx.x >> 5;
    const int lane = threadIdx.x & 31;
    const int half = lane >> 4;                    // lane group 0/1
    const int lm   = lane & 15;
    const int m0   = blockIdx.x * 128 + wave * 16; // wave-uniform -> EXEC stays all-1s
    if (m0 >= nrows) return;

    v8f acc[8];
#pragma unroll
    for (int t = 0; t < 8; ++t) acc[t] = (v8f){0.f,0.f,0.f,0.f,0.f,0.f,0.f,0.f};

    const float* arow = H + (size_t)(m0 + lm) * DIM;

#pragma unroll
    for (int kc = 0; kc < 4; ++kc) {               // K chunks of 32
        const float* ap = arow + kc * 32;
        // A fragment, this code's contiguous k order: e=0..15 -> K = 16*half + e
        float av[16];
        *(float4*)(av + 0)  = *(const float4*)(ap + 16 * half);
        *(float4*)(av + 4)  = *(const float4*)(ap + 16 * half + 4);
        *(float4*)(av + 8)  = *(const float4*)(ap + 16 * half + 8);
        *(float4*)(av + 12) = *(const float4*)(ap + 16 * half + 12);
        v16bf af;
#pragma unroll
        for (int e = 0; e < 16; ++e) af[e] = (__bf16)av[e];

        // B fragment: lane holds column n = nt*16 + lm, K = kc*32 + 16*half + e
        const __bf16* bbase = &WT[kc * 32 + 16 * half];
#pragma unroll
        for (int nt = 0; nt < 8; ++nt) {
            v16bf bf = *(const v16bf*)(bbase + (nt * 16 + lm) * DIM);
            acc[nt] = __builtin_amdgcn_wmma_f32_16x16x32_bf16(
                false, af, false, bf, (short)0, acc[nt], false, false);
        }
    }

    // C/D layout: lane n = lm, VGPR r -> row m0 + 8*half + r
    const int mbase = m0 + 8 * half;
#pragma unroll
    for (int nt = 0; nt < 8; ++nt) {
        float* o = OUT + (size_t)mbase * DIM + nt * 16 + lm;
#pragma unroll
        for (int r = 0; r < 8; ++r) o[(size_t)r * DIM] = acc[nt][r];
    }
}

// ---------------------------------------------------------------------------
// One wave per edge: AGG[col,:] += HW[row,:] * (dinv[row]*ew*dinv[col])
// 4 floats per lane (128 / 32). Non-returning f32 atomics.
// ---------------------------------------------------------------------------
__global__ __launch_bounds__(256) void scatter_kernel(const long long* __restrict__ rowi,
                                                      const long long* __restrict__ coli,
                                                      const float* __restrict__ ew,
                                                      const float* __restrict__ dinv,
                                                      const float* __restrict__ HW,
                                                      float* __restrict__ AGG,
                                                      int nedges) {
    const int wave = threadIdx.x >> 5;
    const int lane = threadIdx.x & 31;
    const long long e = (long long)blockIdx.x * 8 + wave;
    if (e >= nedges) return;
    const int r = (int)rowi[e];
    const int c = (int)coli[e];
    const float norm = dinv[r] * ew[e] * dinv[c];

    const float4 v = *(const float4*)(HW + (size_t)r * DIM + lane * 4);
    float* dst = AGG + (size_t)c * DIM + lane * 4;
    atomicAdd(dst + 0, v.x * norm);
    atomicAdd(dst + 1, v.y * norm);
    atomicAdd(dst + 2, v.z * norm);
    atomicAdd(dst + 3, v.w * norm);
}

// ---------------------------------------------------------------------------
// In place: AGG = relu(AGG + HW * dinv^2 (self-loop) + bias)
// ---------------------------------------------------------------------------
__global__ void finalize_kernel(float* __restrict__ AGG,
                                const float* __restrict__ HW,
                                const float* __restrict__ dinv,
                                const float* __restrict__ bias,
                                int total) {
    int i = blockIdx.x * blockDim.x + threadIdx.x;
    if (i >= total) return;
    const int node = i >> 7;
    const int d    = i & 127;
    const float di = dinv[node];
    const float v  = AGG[i] + HW[i] * di * di + bias[d];
    AGG[i] = fmaxf(v, 0.0f);
}

// ---------------------------------------------------------------------------
extern "C" void kernel_launch(void* const* d_in, const int* in_sizes, int n_in,
                              void* d_out, int out_size, void* d_ws, size_t ws_size,
                              hipStream_t stream) {
    const float*     x   = (const float*)d_in[0];
    const long long* ei  = (const long long*)d_in[1];   // int64 (2, E)
    const float*     ew  = (const float*)d_in[2];
    const float*     W1  = (const float*)d_in[3];
    const float*     b1  = (const float*)d_in[4];
    const float*     W2  = (const float*)d_in[5];
    const float*     b2  = (const float*)d_in[6];
    float*           out = (float*)d_out;

    const int nnodes = in_sizes[0] / DIM;
    const int nedges = in_sizes[2];
    const long long* rowi = ei;
    const long long* coli = ei + nedges;

    // workspace: [dinv | hw | z]
    float* ws = (float*)d_ws;
    const size_t npad = ((size_t)nnodes + 255) & ~(size_t)255;
    float* dinv = ws;
    float* hw   = ws + npad;
    float* z    = hw + (size_t)nnodes * DIM;

    const int T = 256;
    const int nd = nnodes * DIM;
    const int nblk  = (nnodes + T - 1) / T;
    const int eblk  = (nedges + T - 1) / T;
    const int ndblk = (nd + T - 1) / T;
    const int mblk  = (nnodes + 127) / 128;
    const int sblk  = (nedges + 7) / 8;

    // normalization: deg -> dinv (in place)
    init_deg_kernel<<<nblk, T, 0, stream>>>(dinv, nnodes);
    degree_kernel<<<eblk, T, 0, stream>>>(coli, ew, dinv, nedges);
    dinv_kernel<<<nblk, T, 0, stream>>>(dinv, nnodes);

    // ---- layer 1: hw = x@W1 ; z = relu(scatter(hw) + hw*dinv^2 + b1) ----
    zero_kernel<<<ndblk, T, 0, stream>>>(z, nd);
    gemm_bf16_kernel<<<mblk, 256, 0, stream>>>(x, W1, hw, nnodes);
    scatter_kernel<<<sblk, 256, 0, stream>>>(rowi, coli, ew, dinv, hw, z, nedges);
    finalize_kernel<<<ndblk, T, 0, stream>>>(z, hw, dinv, b1, nd);

    // ---- layer 2: hw = z@W2 ; out = relu(scatter(hw) + hw*dinv^2 + b2) ----
    zero_kernel<<<ndblk, T, 0, stream>>>(out, nd);
    gemm_bf16_kernel<<<mblk, 256, 0, stream>>>(z, W2, hw, nnodes);
    scatter_kernel<<<sblk, 256, 0, stream>>>(rowi, coli, ew, dinv, hw, out, nedges);
    finalize_kernel<<<ndblk, T, 0, stream>>>(out, hw, dinv, b2, nd);
}
